// PolicyHead_28518582846074
// MI455X (gfx1250) — matrix operand order repaired and checked by reading, not compiled, not run on hardware
//
#include <hip/hip_runtime.h>
#include <hip/hip_bf16.h>
#include <stdint.h>

// ---------------------------------------------------------------------------
// PolicyHead for MI455X (gfx1250): wave32 + v_wmma_f32_16x16x32_f16.
// One workgroup (8 wave32) per (t,b); all dense matmuls (10x256 padded to
// 16x256) run on WMMA with f16 operands / f32 accumulation.
// Round 2: fused K-loop sharing the A fragment across both N tiles per wave
// (halves ds_load traffic, back-to-back independent v_wmma), plus
// global_prefetch of the per-(t,b) embeddings at kernel entry.
// ---------------------------------------------------------------------------

#define T_DIM 32
#define B_DIM 256
#define TB    (T_DIM * B_DIM)          // 8192
#define A_DIM 10
#define E_DIM 256
#define S_DIM 512
#define MF    128                      // MOVE_FEAT
#define NM    354                      // NUM_MOVES
#define LDA   520                      // f16 LDS row stride (halfs), 16B-skewed

typedef _Float16 v16h __attribute__((ext_vector_type(16)));
typedef _Float16 h8   __attribute__((ext_vector_type(8)));
typedef float    v8f  __attribute__((ext_vector_type(8)));

struct Args {
  const float*         active_weight;    // (T,B,1,6)
  const int*           teams;            // (T,B,8,2,6,48) int32
  const unsigned char* legal;            // (T,B,10) bool8
  const float*         active_embedding; // (T,B,1,E)
  const float*         entity_embeddings;// (T,B,6,E)
  const float*         state_embedding;  // (T,B,S)
  const float*         moves_table;      // (NM,MF)
  const float*         legal_table;      // (2,E)
  const float*         mg_b[3];
  const float*         mg_gw[3];
  const float*         mg_gb[3];
  const float*         bq; const float* bk; const float* bv; const float* bo;
  const float*         pq_w; const float* pq_b;
  const float*         pk_b[3];
  const _Float16*      w_stream[3];      // transposed f16 [N=256][K]
  const _Float16*      wq; const _Float16* wk; const _Float16* wv; const _Float16* wo;
  const _Float16*      pk[3];
  float*               out;              // policy | log_policy | logits
};

// f32 (K x 256 row-major) -> f16 transposed [n*K + k]
__global__ void transpose_to_f16(const float* __restrict__ src,
                                 _Float16* __restrict__ dst, int K) {
  int idx = blockIdx.x * blockDim.x + threadIdx.x;
  if (idx >= K * 256) return;
  int n = idx / K, k = idx - n * K;
  dst[idx] = (_Float16)src[(size_t)k * 256 + n];
}

__device__ __forceinline__ float wredf(float x) {
  #pragma unroll
  for (int off = 16; off; off >>= 1) x += __shfl_xor(x, off, 32);
  return x;
}

// D(16x256) = A(16xK, f16 in LDS, row stride lda) @ Wt(f16, [n][k]) + bias.
// Each wave owns two adjacent 16-wide N tiles; one fused K-loop shares the
// A fragment between both tiles (two independent accumulators).
__device__ __forceinline__ void wmma_mm(const _Float16* lA, int lda,
                                        const _Float16* __restrict__ Wt, int K,
                                        const float* __restrict__ bias,
                                        float* out, int ldo) {
  const int tid  = threadIdx.x;
  const int lane = tid & 31;
  const int wave = tid >> 5;
  const int r16  = lane & 15;
  const int hi   = lane >> 4;
  const int nn0  = wave << 5;                       // first of two N tiles
  const _Float16* ap  = lA + r16 * lda + hi * 8;    // A: M=r16, K split by hi
  const _Float16* wp0 = Wt + (size_t)(nn0 + r16) * K + hi * 8;         // N tile 0
  const _Float16* wp1 = Wt + (size_t)(nn0 + 16 + r16) * K + hi * 8;    // N tile 1
  v8f c0 = {}, c1 = {};
  for (int kk = 0; kk < K; kk += 32) {
    h8 a0  = *(const h8*)(ap + kk);         // K kk+hi*8    .. +7
    h8 a1  = *(const h8*)(ap + kk + 16);    // K kk+16+hi*8 .. +7
    h8 b00 = *(const h8*)(wp0 + kk);
    h8 b01 = *(const h8*)(wp0 + kk + 16);
    h8 b10 = *(const h8*)(wp1 + kk);
    h8 b11 = *(const h8*)(wp1 + kk + 16);
    v16h av, bv0, bv1;
    #pragma unroll
    for (int i = 0; i < 8; ++i) {
      av[i]  = a0[i];  av[i + 8]  = a1[i];
      bv0[i] = b00[i]; bv0[i + 8] = b01[i];
      bv1[i] = b10[i]; bv1[i + 8] = b11[i];
    }
    c0 = __builtin_amdgcn_wmma_f32_16x16x32_f16(false, av, false, bv0,
                                                (short)0, c0, false, false);
    c1 = __builtin_amdgcn_wmma_f32_16x16x32_f16(false, av, false, bv1,
                                                (short)0, c1, false, false);
  }
  const float bb0 = bias[nn0 + r16];
  const float bb1 = bias[nn0 + 16 + r16];
  #pragma unroll
  for (int v = 0; v < 8; ++v) {                      // C/D layout: M=v+8*hi
    out[(v + 8 * hi) * ldo + nn0 + r16]      = c0[v] + bb0;
    out[(v + 8 * hi) * ldo + nn0 + 16 + r16] = c1[v] + bb1;
  }
}

__global__ __launch_bounds__(256) void policy_head_kernel(Args a) {
  __shared__ __align__(16) _Float16 h16[16 * LDA];   // f16 activation staging
  __shared__ float obuf[3][16 * 256];                // stream outs, later q/k/v
  __shared__ float xbuf[16 * 256];                   // x / kv chain
  __shared__ float gates[16][4];
  __shared__ float gvals[16][4];
  __shared__ float sc[2][10][12];
  __shared__ int   tok[16];
  __shared__ float sh[S_DIM];
  __shared__ float qvv[E_DIM];
  __shared__ float lg[16];

  const int tid  = threadIdx.x;
  const int lane = tid & 31;
  const int wave = tid >> 5;
  const int tb   = blockIdx.x;

  // ---- prefetch this block's unique embeddings (HBM latency overlap).
  // 72 cachelines of 128B: active(8) + entity(48) + state(16).
  if (tid < 72) {
    const char* p;
    if (tid < 8)
      p = (const char*)(a.active_embedding + (size_t)tb * E_DIM) + tid * 128;
    else if (tid < 56)
      p = (const char*)(a.entity_embeddings + (size_t)tb * 6 * E_DIM) + (tid - 8) * 128;
    else
      p = (const char*)(a.state_embedding + (size_t)tb * S_DIM) + (tid - 56) * 128;
    __builtin_prefetch(p, 0, 1);                    // -> global_prefetch_b8
  }

  // ---- stage 0: action tokens (one-hot select of teams[...,7,0,:,44:48])
  if (tid < A_DIM) {
    int t = 0;                                   // SWITCH_TOKEN
    if (tid < 4) {
      const int* base = a.teams + ((size_t)tb * 8 + 7) * (2 * 6 * 48);
      float s = 0.f;
      #pragma unroll
      for (int j = 0; j < 6; ++j)
        s += a.active_weight[(size_t)tb * 6 + j] * (float)base[j * 48 + 44 + tid];
      t = (int)(s + 0.5f);
      t = t < 0 ? 0 : (t >= NM ? NM - 1 : t);
    }
    tok[tid] = t;
  }
  if (tid < 16) { gates[tid][0] = 0.f; gates[tid][1] = 0.f; gates[tid][2] = 0.f; }
  __syncthreads();

  // ---- stage 1: three gated streams  h = relu(ln(feat)); out = h@W+b
  for (int sid = 0; sid < 3; ++sid) {
    const int sz = (sid == 0) ? MF : E_DIM;
    for (int idx = tid; idx < 6 * sz; idx += 256)          // zero pad rows 10..15
      h16[(10 + idx / sz) * LDA + (idx % sz)] = (_Float16)0.f;
    for (int r = wave; r < A_DIM; r += 8) {
      const int nk = sz >> 5;
      float v[8]; float s = 0.f, s2 = 0.f;
      for (int i = 0; i < nk; ++i) {
        const int k = lane + (i << 5);
        float x;
        if (sid == 0)       x = a.moves_table[(size_t)tok[r] * MF + k];
        else if (sid == 1)  x = (r < 4) ? a.active_embedding[(size_t)tb * E_DIM + k]
                                        : a.entity_embeddings[((size_t)tb * 6 + (r - 4)) * E_DIM + k];
        else {
          const int l = a.legal[(size_t)tb * A_DIM + r] ? 1 : 0;
          x = a.legal_table[l * E_DIM + k];
        }
        v[i] = x; s += x; s2 += x * x;
      }
      s = wredf(s); s2 = wredf(s2);
      const float m   = s / (float)sz;
      const float inv = rsqrtf(fmaxf(s2 / (float)sz - m * m, 0.f) + 1e-5f);
      const float* gw = a.mg_gw[sid];
      float g0 = 0.f, g1 = 0.f, g2 = 0.f;
      for (int i = 0; i < nk; ++i) {
        const int k = lane + (i << 5);
        float h = fmaxf((v[i] - m) * inv, 0.f);
        h16[r * LDA + k] = (_Float16)h;
        g0 += h * gw[k * 3 + 0]; g1 += h * gw[k * 3 + 1]; g2 += h * gw[k * 3 + 2];
      }
      g0 = wredf(g0); g1 = wredf(g1); g2 = wredf(g2);
      if (lane == 0) {
        const float* gb = a.mg_gb[sid];
        gates[r][0] += g0 + gb[0]; gates[r][1] += g1 + gb[1]; gates[r][2] += g2 + gb[2];
      }
    }
    __syncthreads();
    wmma_mm(h16, LDA, a.w_stream[sid], sz, a.mg_b[sid], &obuf[sid][0], 256);
    __syncthreads();
  }

  // ---- gate softmax + mixed x (also pre-stage f16 x with zero pad rows)
  if (tid < A_DIM) {
    float g0 = gates[tid][0], g1 = gates[tid][1], g2 = gates[tid][2];
    float mx = fmaxf(g0, fmaxf(g1, g2));
    float e0 = __expf(g0 - mx), e1 = __expf(g1 - mx), e2 = __expf(g2 - mx);
    float inv = 1.f / (e0 + e1 + e2);
    gvals[tid][0] = e0 * inv; gvals[tid][1] = e1 * inv; gvals[tid][2] = e2 * inv;
  }
  __syncthreads();
  for (int idx = tid; idx < 16 * 256; idx += 256) {
    const int r = idx >> 8;
    float x = 0.f;
    if (r < A_DIM)
      x = gvals[r][0] * obuf[0][idx] + gvals[r][1] * obuf[1][idx] + gvals[r][2] * obuf[2][idx];
    xbuf[idx] = x;
    h16[r * LDA + (idx & 255)] = (_Float16)x;
  }
  __syncthreads();

  // ---- stage 2: attention  q/k/v projections on WMMA
  wmma_mm(h16, LDA, a.wq, E_DIM, a.bq, &obuf[0][0], 256);
  wmma_mm(h16, LDA, a.wk, E_DIM, a.bk, &obuf[1][0], 256);
  wmma_mm(h16, LDA, a.wv, E_DIM, a.bv, &obuf[2][0], 256);
  __syncthreads();

  if (tid < 200) {                                 // scores: 2 heads x 10 x 10
    const int hh = tid / 100, qa = (tid / 10) % 10, ka = tid % 10;
    const float* qp = &obuf[0][qa * 256 + hh * 128];
    const float* kp = &obuf[1][ka * 256 + hh * 128];
    float s = 0.f;
    for (int d = 0; d < 128; ++d) s += qp[d] * kp[d];
    sc[hh][qa][ka] = s * 0.08838834764831845f;     // 1/sqrt(128)
  }
  __syncthreads();
  if (tid < 20) {                                  // softmax over k
    const int hh = tid / 10, qa = tid % 10;
    float mx = -1e30f;
    for (int k2 = 0; k2 < 10; ++k2) mx = fmaxf(mx, sc[hh][qa][k2]);
    float e[10], sum = 0.f;
    for (int k2 = 0; k2 < 10; ++k2) { e[k2] = __expf(sc[hh][qa][k2] - mx); sum += e[k2]; }
    const float inv = 1.f / sum;
    for (int k2 = 0; k2 < 10; ++k2) sc[hh][qa][k2] = e[k2] * inv;
  }
  __syncthreads();
  {                                                // o = att @ v -> f16 staging
    const int e0 = tid, hh = e0 >> 7;
    for (int a2 = 0; a2 < A_DIM; ++a2) {
      float s = 0.f;
      for (int k2 = 0; k2 < 10; ++k2) s += sc[hh][a2][k2] * obuf[2][k2 * 256 + e0];
      h16[a2 * LDA + e0] = (_Float16)s;            // rows 10..15 stay zero
    }
  }
  __syncthreads();
  wmma_mm(h16, LDA, a.wo, E_DIM, a.bo, xbuf, 256);
  __syncthreads();

  // ---- state query qv = relu(ln(state)) @ pq_w + pq_b  (M=1 -> VALU)
  if (wave == 0) {
    float v[16]; float s = 0.f, s2 = 0.f;
    #pragma unroll
    for (int i = 0; i < 16; ++i) {
      float x = a.state_embedding[(size_t)tb * S_DIM + lane + (i << 5)];
      v[i] = x; s += x; s2 += x * x;
    }
    s = wredf(s); s2 = wredf(s2);
    const float m   = s * (1.f / S_DIM);
    const float inv = rsqrtf(fmaxf(s2 * (1.f / S_DIM) - m * m, 0.f) + 1e-5f);
    #pragma unroll
    for (int i = 0; i < 16; ++i) sh[lane + (i << 5)] = fmaxf((v[i] - m) * inv, 0.f);
  }
  __syncthreads();
  {
    float s = a.pq_b[tid];
    for (int k = 0; k < S_DIM; ++k) s += sh[k] * a.pq_w[(size_t)k * E_DIM + tid];
    qvv[tid] = s;
  }
  __syncthreads();

  // ---- stage 3: kv chain  kv = relu(ln(kv)) @ pk_j + b, j = 0..2
  for (int j = 0; j < 3; ++j) {
    for (int r = wave; r < A_DIM; r += 8) {
      float v[8]; float s = 0.f, s2 = 0.f;
      #pragma unroll
      for (int i = 0; i < 8; ++i) {
        float x = xbuf[r * 256 + lane + (i << 5)];
        v[i] = x; s += x; s2 += x * x;
      }
      s = wredf(s); s2 = wredf(s2);
      const float m   = s * (1.f / 256.f);
      const float inv = rsqrtf(fmaxf(s2 * (1.f / 256.f) - m * m, 0.f) + 1e-5f);
      #pragma unroll
      for (int i = 0; i < 8; ++i)
        h16[r * LDA + lane + (i << 5)] = (_Float16)fmaxf((v[i] - m) * inv, 0.f);
    }
    __syncthreads();
    wmma_mm(h16, LDA, a.pk[j], E_DIM, a.pk_b[j], xbuf, 256);
    __syncthreads();
  }

  // ---- stage 4: logits + masked softmax / log-softmax
  if (tid < A_DIM) {
    float s = 0.f;
    for (int k = 0; k < 256; ++k) s += qvv[k] * xbuf[tid * 256 + k];
    lg[tid] = s;
  }
  __syncthreads();
  if (tid == 0) {
    float L[10]; bool leg[10];
    float lmaxLegal = -1e30f;
    for (int i = 0; i < A_DIM; ++i) {
      L[i]  = lg[i];
      leg[i] = a.legal[(size_t)tb * A_DIM + i] != 0;
      if (leg[i]) lmaxLegal = fmaxf(lmaxLegal, L[i]);
    }
    float ex[10], sum = 0.f;
    for (int i = 0; i < A_DIM; ++i) {
      ex[i] = leg[i] ? __expf(L[i] - lmaxLegal) : 0.f;
      sum += ex[i];
    }
    const float inv      = 1.f / sum;
    const float baseline = __logf(sum) + lmaxLegal;   // logsumexp over legal
    float* po = a.out;
    for (int i = 0; i < A_DIM; ++i) {
      const size_t o = (size_t)tb * A_DIM + i;
      po[o]                          = ex[i] * inv;                      // policy
      po[(size_t)TB * A_DIM + o]     = leg[i] ? (L[i] - baseline) : 0.f; // log_policy
      po[(size_t)2 * TB * A_DIM + o] = L[i];                             // logits
    }
  }
}

// ---------------------------------------------------------------------------
extern "C" void kernel_launch(void* const* d_in, const int* in_sizes, int n_in,
                              void* d_out, int out_size, void* d_ws, size_t ws_size,
                              hipStream_t stream) {
  (void)in_sizes; (void)n_in; (void)out_size; (void)ws_size;

  // d_in order: active_weight, teams, legal, active_embedding,
  // entity_embeddings, state_embedding, then params in dict-insertion order.
  const float* mg_w[3]  = { (const float*)d_in[8],  (const float*)d_in[12], (const float*)d_in[16] };
  const float* wq_f     = (const float*)d_in[20];
  const float* wk_f     = (const float*)d_in[22];
  const float* wv_f     = (const float*)d_in[24];
  const float* wo_f     = (const float*)d_in[26];
  const float* pk_f[3]  = { (const float*)d_in[30], (const float*)d_in[32], (const float*)d_in[34] };

  _Float16* ws = (_Float16*)d_ws;
  // f16 transposed weight arena (halfs): 256 * (128 + 9*256) = 622,592 (~1.2 MB)
  size_t off[10];
  size_t cur = 0;
  const int Ks[10]       = { MF, E_DIM, E_DIM, E_DIM, E_DIM, E_DIM, E_DIM, E_DIM, E_DIM, E_DIM };
  const float* srcs[10]  = { mg_w[0], mg_w[1], mg_w[2], wq_f, wk_f, wv_f, wo_f,
                             pk_f[0], pk_f[1], pk_f[2] };
  for (int i = 0; i < 10; ++i) { off[i] = cur; cur += (size_t)Ks[i] * 256; }
  for (int i = 0; i < 10; ++i) {
    const int total = Ks[i] * 256;
    transpose_to_f16<<<(total + 255) / 256, 256, 0, stream>>>(srcs[i], ws + off[i], Ks[i]);
  }

  Args a;
  a.active_weight     = (const float*)d_in[0];
  a.teams             = (const int*)d_in[1];
  a.legal             = (const unsigned char*)d_in[2];
  a.active_embedding  = (const float*)d_in[3];
  a.entity_embeddings = (const float*)d_in[4];
  a.state_embedding   = (const float*)d_in[5];
  a.moves_table       = (const float*)d_in[6];
  a.legal_table       = (const float*)d_in[7];
  a.mg_b[0]  = (const float*)d_in[9];  a.mg_gw[0] = (const float*)d_in[10]; a.mg_gb[0] = (const float*)d_in[11];
  a.mg_b[1]  = (const float*)d_in[13]; a.mg_gw[1] = (const float*)d_in[14]; a.mg_gb[1] = (const float*)d_in[15];
  a.mg_b[2]  = (const float*)d_in[17]; a.mg_gw[2] = (const float*)d_in[18]; a.mg_gb[2] = (const float*)d_in[19];
  a.bq = (const float*)d_in[21]; a.bk = (const float*)d_in[23];
  a.bv = (const float*)d_in[25]; a.bo = (const float*)d_in[27];
  a.pq_w = (const float*)d_in[28]; a.pq_b = (const float*)d_in[29];
  a.pk_b[0] = (const float*)d_in[31]; a.pk_b[1] = (const float*)d_in[33]; a.pk_b[2] = (const float*)d_in[35];
  a.w_stream[0] = ws + off[0]; a.w_stream[1] = ws + off[1]; a.w_stream[2] = ws + off[2];
  a.wq = ws + off[3]; a.wk = ws + off[4]; a.wv = ws + off[5]; a.wo = ws + off[6];
  a.pk[0] = ws + off[7]; a.pk[1] = ws + off[8]; a.pk[2] = ws + off[9];
  a.out = (float*)d_out;

  policy_head_kernel<<<TB, 256, 0, stream>>>(a);
}
